// FusedMoEModule_78228534329620
// MI455X (gfx1250) — matrix operand order repaired and checked by reading, not compiled, not run on hardware
//
#include <hip/hip_runtime.h>

#define T_TOK   8192
#define D_MODEL 2048
#define HID     5632
#define N_EXP   8
#define N_PAIR  16384   // T_TOK * TOP_K

typedef __bf16 bf16_t;
typedef bf16_t v16bf __attribute__((ext_vector_type(16)));
typedef bf16_t v8bf  __attribute__((ext_vector_type(8)));
typedef float  v8f   __attribute__((ext_vector_type(8)));
typedef int    v4i   __attribute__((vector_size(16)));

union V8F { v8f v; float f[8]; };

#if defined(__gfx1250__) && __has_builtin(__builtin_amdgcn_global_load_async_to_lds_b128) && \
    __has_builtin(__builtin_amdgcn_s_wait_asynccnt)
#define USE_ASYNC_LDS 1
#else
#define USE_ASYNC_LDS 0
#endif

__device__ __forceinline__ unsigned short f2bf(float f) {
    union { float f; unsigned int u; } x; x.f = f;
    unsigned int r = x.u + 0x7FFFu + ((x.u >> 16) & 1u);
    return (unsigned short)(r >> 16);
}

// 16-byte global -> LDS copy; async DMA when the toolchain exposes it.
__device__ __forceinline__ void copy16_g2l(void* lds, const void* g) {
#if USE_ASYNC_LDS
    __builtin_amdgcn_global_load_async_to_lds_b128(
        (__attribute__((address_space(1))) v4i*)(g),
        (__attribute__((address_space(3))) v4i*)(lds), 0, 0);
#else
    *(uint4*)lds = *(const uint4*)g;
#endif
}

__device__ __forceinline__ void wait_async() {
#if USE_ASYNC_LDS
    __builtin_amdgcn_s_wait_asynccnt(0);
#endif
}

// A fragment (16-bit A, 16x32): lane<16 -> row m=lane, K in {kk..kk+7} U {kk+16..kk+23};
// lane>=16 -> row m=lane-16, K shifted by +8.
__device__ __forceinline__ v16bf load_frag_a(const unsigned short* s, int rbase, int kk,
                                             int pitch, int lane) {
    int r  = rbase + (lane & 15);
    int h8 = (lane >> 4) << 3;
    const unsigned short* p = s + r * pitch + kk + h8;
    union { v16bf v; v8bf h[2]; } u;
    u.h[0] = *(const v8bf*)(p);
    u.h[1] = *(const v8bf*)(p + 16);
    return u.v;
}

// B fragment (16-bit B, 32x16): lane<16 -> col n=lane, K in {kk..kk+7} U {kk+16..kk+23};
// lane>=16 -> col n=lane-16, K shifted by +8.
__device__ __forceinline__ v16bf load_frag_b(const unsigned short* s, int nbase, int kk,
                                             int pitch, int lane) {
    int r   = nbase + (lane & 15);
    int h16 = (lane >> 4) << 3;
    const unsigned short* p = s + r * pitch + kk + h16;
    union { v16bf v; v8bf h[2]; } u;
    u.h[0] = *(const v8bf*)(p);
    u.h[1] = *(const v8bf*)(p + 16);
    return u.v;
}

__device__ __forceinline__ v8f wmma_bf16(v16bf a, v16bf b, v8f c) {
    return __builtin_amdgcn_wmma_f32_16x16x32_bf16(false, a, false, b, (short)0, c,
                                                   false, false);
}

// ---------------------------------------------------------------- zero + routing
__global__ __launch_bounds__(256) void zero_init(float* __restrict__ out,
                                                 int* __restrict__ cnt) {
    int idx = blockIdx.x * 256 + threadIdx.x;           // 16384 blocks -> exact cover
    ((float4*)out)[idx] = make_float4(0.f, 0.f, 0.f, 0.f);
    if (blockIdx.x == 0 && threadIdx.x < N_EXP) cnt[threadIdx.x] = 0;
}

__global__ __launch_bounds__(256) void route_pairs(const int* __restrict__ ids,
                                                   const float* __restrict__ w,
                                                   int* __restrict__ cnt,
                                                   int* __restrict__ tIds,
                                                   float* __restrict__ wts) {
    int p = blockIdx.x * 256 + threadIdx.x;
    if (p >= N_PAIR) return;
    int e = ids[p];
    int slot = atomicAdd(&cnt[e], 1);
    tIds[e * N_PAIR + slot] = p >> 1;   // token id
    wts [e * N_PAIR + slot] = w[p];     // routed weight
}

__global__ void scan_counts(const int* __restrict__ cnt, int* __restrict__ base) {
    int acc = 0;
    for (int e = 0; e < N_EXP; ++e) { base[e] = acc; acc += cnt[e]; }
}

// ---------------------------------------------------------------- GEMM1: X @ Wup^T, SwiGLU
// grid = (44 = HID/128, 128 pair tiles, 8 experts), block = 256 (8 waves)
__global__ __launch_bounds__(256) void moe_gemm1(const float* __restrict__ xp,
                                                 const float* __restrict__ upw,
                                                 const int* __restrict__ cnt,
                                                 const int* __restrict__ basep,
                                                 const int* __restrict__ tIds,
                                                 unsigned short* __restrict__ act) {
    const int e  = blockIdx.z;
    const int cn = cnt[e];
    const int ty = blockIdx.y;
    if (ty * 128 >= cn) return;                 // uniform early-out: EXEC stays all-ones
    const int nt   = blockIdx.x;
    const int tid  = threadIdx.x;
    const int lane = tid & 31;
    const int wv   = tid >> 5;
    const int wr   = wv >> 1;                   // 0..3 : 32-row strip
    const int wc   = wv & 1;                    // 0..1 : 64-col strip

    __shared__ unsigned short Xs[128 * 72];
    __shared__ unsigned short Gs[128 * 72];
    __shared__ unsigned short Us[128 * 72];
    __shared__ int toks[128];

    if (tid < 128) {
        int i = ty * 128 + tid;
        int cl = i < cn ? i : cn - 1;
        toks[tid] = tIds[e * N_PAIR + cl];
    }
    __syncthreads();

    const float* Wg = upw + (size_t)e * (2 * HID) * D_MODEL + (size_t)(nt * 128) * D_MODEL;
    const float* Wu = Wg + (size_t)HID * D_MODEL;

    v8f cg[2][4], cu[2][4];
#pragma unroll
    for (int i = 0; i < 2; ++i)
#pragma unroll
        for (int j = 0; j < 4; ++j) {
            cg[i][j] = (v8f){0.f, 0.f, 0.f, 0.f, 0.f, 0.f, 0.f, 0.f};
            cu[i][j] = (v8f){0.f, 0.f, 0.f, 0.f, 0.f, 0.f, 0.f, 0.f};
        }

    for (int k0 = 0; k0 < D_MODEL; k0 += 64) {
        // stage X (gathered rows), fp32 -> bf16
#pragma unroll
        for (int i = 0; i < 8; ++i) {
            int c = i * 256 + tid, row = c >> 4, col = (c & 15) << 2;
            float4 v = *(const float4*)(xp + (size_t)toks[row] * D_MODEL + k0 + col);
            unsigned int lo = (unsigned int)f2bf(v.x) | ((unsigned int)f2bf(v.y) << 16);
            unsigned int hi = (unsigned int)f2bf(v.z) | ((unsigned int)f2bf(v.w) << 16);
            *(uint2*)&Xs[row * 72 + col] = make_uint2(lo, hi);
        }
        // stage gate / up weight tiles
#pragma unroll
        for (int i = 0; i < 8; ++i) {
            int c = i * 256 + tid, row = c >> 4, col = (c & 15) << 2;
            float4 v = *(const float4*)(Wg + (size_t)row * D_MODEL + k0 + col);
            unsigned int lo = (unsigned int)f2bf(v.x) | ((unsigned int)f2bf(v.y) << 16);
            unsigned int hi = (unsigned int)f2bf(v.z) | ((unsigned int)f2bf(v.w) << 16);
            *(uint2*)&Gs[row * 72 + col] = make_uint2(lo, hi);
        }
#pragma unroll
        for (int i = 0; i < 8; ++i) {
            int c = i * 256 + tid, row = c >> 4, col = (c & 15) << 2;
            float4 v = *(const float4*)(Wu + (size_t)row * D_MODEL + k0 + col);
            unsigned int lo = (unsigned int)f2bf(v.x) | ((unsigned int)f2bf(v.y) << 16);
            unsigned int hi = (unsigned int)f2bf(v.z) | ((unsigned int)f2bf(v.w) << 16);
            *(uint2*)&Us[row * 72 + col] = make_uint2(lo, hi);
        }
        __syncthreads();
#pragma unroll
        for (int kk = 0; kk < 64; kk += 32) {
            v16bf a0 = load_frag_a(Xs, wr * 32, kk, 72, lane);
            v16bf a1 = load_frag_a(Xs, wr * 32 + 16, kk, 72, lane);
#pragma unroll
            for (int j = 0; j < 4; ++j) {
                v16bf b = load_frag_b(Gs, wc * 64 + j * 16, kk, 72, lane);
                cg[0][j] = wmma_bf16(a0, b, cg[0][j]);
                cg[1][j] = wmma_bf16(a1, b, cg[1][j]);
            }
#pragma unroll
            for (int j = 0; j < 4; ++j) {
                v16bf b = load_frag_b(Us, wc * 64 + j * 16, kk, 72, lane);
                cu[0][j] = wmma_bf16(a0, b, cu[0][j]);
                cu[1][j] = wmma_bf16(a1, b, cu[1][j]);
            }
        }
        __syncthreads();
    }

    // SwiGLU + store act (bf16) to workspace
    const int be = basep[e];
    const int mh = (lane >> 4) << 3;
    const int nl = lane & 15;
#pragma unroll
    for (int i = 0; i < 2; ++i)
#pragma unroll
        for (int j = 0; j < 4; ++j) {
            V8F g, u; g.v = cg[i][j]; u.v = cu[i][j];
            int gcol = nt * 128 + wc * 64 + j * 16 + nl;
#pragma unroll
            for (int q = 0; q < 8; ++q) {
                int r = wr * 32 + i * 16 + q + mh;
                if (ty * 128 + r < cn) {
                    float gf = g.f[q];
                    float a  = (gf / (1.0f + __expf(-gf))) * u.f[q];
                    act[(size_t)(be + ty * 128 + r) * HID + gcol] = f2bf(a);
                }
            }
        }
}

// ---------------------------------------------------------------- GEMM2: act @ Wdown^T, combine
// Double-buffered, async A staging (act is already bf16). K-step = 32, pitch 40.
// grid = (16 = D/128, 128 pair tiles, 8 experts), block = 256 (8 waves)
#define G2_PITCH 40
#define G2_BUF   (128 * G2_PITCH)   // ushorts per buffer

__global__ __launch_bounds__(256) void moe_gemm2(const unsigned short* __restrict__ act,
                                                 const float* __restrict__ dww,
                                                 const int* __restrict__ cnt,
                                                 const int* __restrict__ basep,
                                                 const int* __restrict__ tIds,
                                                 const float* __restrict__ wts,
                                                 float* __restrict__ out) {
    const int e  = blockIdx.z;
    const int cn = cnt[e];
    const int ty = blockIdx.y;
    if (ty * 128 >= cn) return;
    const int dt   = blockIdx.x;
    const int tid  = threadIdx.x;
    const int lane = tid & 31;
    const int wv   = tid >> 5;
    const int wr   = wv >> 1;
    const int wc   = wv & 1;

    __shared__ unsigned short As[2 * G2_BUF];
    __shared__ unsigned short Bs[2 * G2_BUF];
    __shared__ int   toks[128];
    __shared__ float wgt[128];

    if (tid < 128) {
        int i = ty * 128 + tid;
        int cl = i < cn ? i : cn - 1;
        toks[tid] = tIds[e * N_PAIR + cl];
        wgt[tid]  = (i < cn) ? wts[e * N_PAIR + cl] : 0.f;
    }

    const int be = basep[e];
    const float* Wd = dww + (size_t)e * D_MODEL * HID + (size_t)(dt * 128) * HID;

    // A tile: 128 rows x 32 K bf16 = 512 x 16B chunks, 2 per thread (async DMA)
    auto stageA = [&](int k0, int buf) {
#pragma unroll
        for (int i = 0; i < 2; ++i) {
            int c = i * 256 + tid, row = c >> 2, col = (c & 3) << 3;
            int rr = ty * 128 + row; rr = rr < cn ? rr : cn - 1;
            copy16_g2l(&As[buf * G2_BUF + row * G2_PITCH + col],
                       act + (size_t)(be + rr) * HID + k0 + col);
        }
    };
    // B tile: 128 rows x 32 K, fp32 -> bf16, 4 float4 per thread
    auto stageB = [&](int k0, int buf) {
#pragma unroll
        for (int i = 0; i < 4; ++i) {
            int c = i * 256 + tid, row = c >> 3, col = (c & 7) << 2;
            float4 v = *(const float4*)(Wd + (size_t)row * HID + k0 + col);
            unsigned int lo = (unsigned int)f2bf(v.x) | ((unsigned int)f2bf(v.y) << 16);
            unsigned int hi = (unsigned int)f2bf(v.z) | ((unsigned int)f2bf(v.w) << 16);
            *(uint2*)&Bs[buf * G2_BUF + row * G2_PITCH + col] = make_uint2(lo, hi);
        }
    };

    v8f acc[2][4];
#pragma unroll
    for (int i = 0; i < 2; ++i)
#pragma unroll
        for (int j = 0; j < 4; ++j)
            acc[i][j] = (v8f){0.f, 0.f, 0.f, 0.f, 0.f, 0.f, 0.f, 0.f};

    // pipeline prologue
    stageA(0, 0);
    stageB(0, 0);
    wait_async();
    __syncthreads();

    int cur = 0;
    for (int k0 = 0; k0 < HID; k0 += 32) {
        const int  nxt  = cur ^ 1;
        const bool more = (k0 + 32) < HID;
        if (more) stageA(k0 + 32, nxt);       // async fills nxt while we compute cur

        v16bf a0 = load_frag_a(&As[cur * G2_BUF], wr * 32, 0, G2_PITCH, lane);
        v16bf a1 = load_frag_a(&As[cur * G2_BUF], wr * 32 + 16, 0, G2_PITCH, lane);
#pragma unroll
        for (int j = 0; j < 4; ++j) {
            v16bf b = load_frag_b(&Bs[cur * G2_BUF], wc * 64 + j * 16, 0, G2_PITCH, lane);
            acc[0][j] = wmma_bf16(a0, b, acc[0][j]);
            acc[1][j] = wmma_bf16(a1, b, acc[1][j]);
        }

        if (more) stageB(k0 + 32, nxt);       // VALU convert overlaps WMMA
        wait_async();
        __syncthreads();
        cur = nxt;
    }

    const int mh = (lane >> 4) << 3;
    const int nl = lane & 15;
#pragma unroll
    for (int i = 0; i < 2; ++i)
#pragma unroll
        for (int j = 0; j < 4; ++j) {
            V8F c; c.v = acc[i][j];
            int col = dt * 128 + wc * 64 + j * 16 + nl;
#pragma unroll
            for (int q = 0; q < 8; ++q) {
                int r = wr * 32 + i * 16 + q + mh;
                if (ty * 128 + r < cn) {
                    float v = c.f[q] * wgt[r];
                    __hip_atomic_fetch_add(&out[(size_t)toks[r] * D_MODEL + col], v,
                                           __ATOMIC_RELAXED, __HIP_MEMORY_SCOPE_AGENT);
                }
            }
        }
}

// ---------------------------------------------------------------- host launcher
extern "C" void kernel_launch(void* const* d_in, const int* in_sizes, int n_in,
                              void* d_out, int out_size, void* d_ws, size_t ws_size,
                              hipStream_t stream) {
    const float* x   = (const float*)d_in[0];  // hidden_states [T, D]
    const float* tw  = (const float*)d_in[1];  // topk_weights  [T, 2]
    const float* upw = (const float*)d_in[2];  // up_weight     [E, 2H, D]
    const float* dww = (const float*)d_in[3];  // down_weight   [E, D, H]
    const int*   ids = (const int*)d_in[4];    // topk_ids      [T, 2]
    float* out = (float*)d_out;

    // workspace layout
    int* cnt   = (int*)d_ws;                                   // [8]
    int* base  = cnt + 8;                                      // [8]
    int* tIds  = cnt + 16;                                     // [8 * N_PAIR]
    float* wts = (float*)(tIds + N_EXP * N_PAIR);              // [8 * N_PAIR]
    unsigned short* act = (unsigned short*)(wts + N_EXP * N_PAIR); // [N_PAIR * HID] bf16

    zero_init<<<(T_TOK * D_MODEL) / (256 * 4), 256, 0, stream>>>(out, cnt);
    route_pairs<<<N_PAIR / 256, 256, 0, stream>>>(ids, tw, cnt, tIds, wts);
    scan_counts<<<1, 1, 0, stream>>>(cnt, base);
    moe_gemm1<<<dim3(HID / 128, N_PAIR / 128, N_EXP), 256, 0, stream>>>(
        x, upw, cnt, base, tIds, act);
    moe_gemm2<<<dim3(D_MODEL / 128, N_PAIR / 128, N_EXP), 256, 0, stream>>>(
        act, dww, cnt, base, tIds, wts, out);
}
